// GraphMERTModel_90288802496731
// MI455X (gfx1250) — matrix operand order repaired and checked by reading, not compiled, not run on hardware
//
#include <hip/hip_runtime.h>
#include <hip/hip_bf16.h>

// ---------------------------------------------------------------------------
// Model dims (reference): B=32,N=512,IN=128,D=256,H=8,L=2,R=10,FF=2048,V=10
// M = B*N = 16384 rows everywhere.  All matmuls via v_wmma_f32_16x16x32_bf16
// (f32 accumulate); softmax/LN/bias/residual in f32.
// Weights are pre-converted to bf16 *transposed* (W^T, [N][K]) so every WMMA
// B-fragment is K-contiguous -> TDM-friendly rectangular tiles + ds_load_b128.
// ---------------------------------------------------------------------------

typedef __bf16 bf16_t;
typedef __attribute__((ext_vector_type(16))) __bf16 v16bf;
typedef __attribute__((ext_vector_type(8)))  float  v8f;
typedef unsigned int v4u __attribute__((ext_vector_type(4)));
typedef int          v8i __attribute__((ext_vector_type(8)));
typedef int          v4i __attribute__((ext_vector_type(4)));

#define WMMA_BF16(a, b, c) \
  __builtin_amdgcn_wmma_f32_16x16x32_bf16(false, (a), false, (b), (short)0, (c), false, false)

union Frag {
  v16bf v;
  uint4 q[2];
  bf16_t e[16];
};

__device__ __forceinline__ v8f vzero8() {
  v8f z;
#pragma unroll
  for (int i = 0; i < 8; ++i) z[i] = 0.f;
  return z;
}

// branch-free tanh-GELU: tanh(u) = sign(u) * (1 - 2/(e^{2|u|}+1))
__device__ __forceinline__ float gelu_f(float x) {
  float u = 0.7978845608028654f * (x + 0.044715f * x * x * x);
  float a = fabsf(u);
  float t = __expf(-2.0f * a);
  float th = (1.0f - t) * __frcp_rn(1.0f + t);
  th = copysignf(th, u);
  return 0.5f * x * (1.0f + th);
}

__device__ __forceinline__ void pack16(Frag& f, float4 a, float4 b, float4 c, float4 d) {
  f.e[0]  = (bf16_t)a.x; f.e[1]  = (bf16_t)a.y; f.e[2]  = (bf16_t)a.z; f.e[3]  = (bf16_t)a.w;
  f.e[4]  = (bf16_t)b.x; f.e[5]  = (bf16_t)b.y; f.e[6]  = (bf16_t)b.z; f.e[7]  = (bf16_t)b.w;
  f.e[8]  = (bf16_t)c.x; f.e[9]  = (bf16_t)c.y; f.e[10] = (bf16_t)c.z; f.e[11] = (bf16_t)c.w;
  f.e[12] = (bf16_t)d.x; f.e[13] = (bf16_t)d.y; f.e[14] = (bf16_t)d.z; f.e[15] = (bf16_t)d.w;
}

// ---------------------------------------------------------------------------
// GEMM: Cout[M x Nn] = postact( preact(A[M x K] f32) @ W + bias + Cin )
// W given transposed bf16: Bt[n*ldb + k]  (ldb = K-stride of W^T row).
// Block tile 64x64, K-step 32, 128 threads = 4 waves, 2x2 WMMA per wave.
// B tile (64 n-rows x 32 k) is loaded into LDS either by the Tensor Data
// Mover (full tiles; wave 0 issues tensor_load_to_lds, s_wait_tensorcnt 0)
// or by vectorized manual loads (partial tiles, e.g. classifier Nn=10).
// ---------------------------------------------------------------------------
__global__ __launch_bounds__(128) void gemm_kernel(
    const float* __restrict__ A, const bf16_t* __restrict__ Bt,
    const float* __restrict__ bias, const float* __restrict__ Cin,
    float* __restrict__ Cout, int M, int Nn, int K, int ldb,
    int preact /*0 none, 2 gelu*/, int postact /*0 none, 1 relu*/,
    int use_tdm) {
  __shared__ __align__(16) bf16_t sA[64][48];  // [m][k], row stride 96B
  __shared__ __align__(16) bf16_t sB[64][32];  // W^T tile: [n][k], packed 64B rows

  const int tid  = threadIdx.x;
  const int wave = tid >> 5;
  const int lane = tid & 31;
  const int g    = lane >> 4;
  const int ln   = lane & 15;
  const int mBase = blockIdx.y << 6;
  const int nBase = blockIdx.x << 6;
  const int wm = (wave >> 1) << 5;
  const int wn = (wave & 1) << 5;

  const bool tdm_ok = (use_tdm != 0) && (nBase + 64 <= Nn);

  v8f acc[2][2];
  acc[0][0] = vzero8(); acc[0][1] = vzero8();
  acc[1][0] = vzero8(); acc[1][1] = vzero8();

  const int ar = tid >> 3;         // 0..15
  const int ac = (tid & 7) << 2;   // 0,4,...,28
  const unsigned ldsB = (unsigned)(uintptr_t)&sB[0][0];  // LDS byte offset (low 32 bits)

  for (int k0 = 0; k0 < K; k0 += 32) {
    // ---- stage A (64x32 f32 -> bf16), 4x float4 per thread -----------------
#pragma unroll
    for (int i = 0; i < 4; ++i) {
      int row = ar + (i << 4);
      const float* ap = A + (size_t)(mBase + row) * K + (k0 + ac);
      float4 f = *(const float4*)ap;
      if (preact == 2) { f.x = gelu_f(f.x); f.y = gelu_f(f.y); f.z = gelu_f(f.z); f.w = gelu_f(f.w); }
      sA[row][ac + 0] = (bf16_t)f.x;
      sA[row][ac + 1] = (bf16_t)f.y;
      sA[row][ac + 2] = (bf16_t)f.z;
      sA[row][ac + 3] = (bf16_t)f.w;
      if (i == 0 && k0 + 32 < K) __builtin_prefetch(ap + 32, 0, 0);
    }
    // ---- stage B tile ------------------------------------------------------
    if (tdm_ok) {
      if (wave == 0) {
        unsigned long long ga =
            (unsigned long long)(uintptr_t)(Bt + (size_t)nBase * ldb + k0);
        v4u g0 = (v4u){1u,                      // count=1 valid descriptor
                       ldsB,                    // lds_addr
                       (unsigned)ga,            // global_addr[31:0]
                       ((unsigned)(ga >> 32) & 0x01FFFFFFu) | 0x80000000u}; // type=2
        v8i g1 = (v8i){(int)(1u << 16),         // data_size = 1 (2 bytes)
                       (int)(32u << 16),        // tensor_dim0 = 32 (k width)
                       (int)(64u << 16),        // tensor_dim1 = 64 (n rows)
                       (int)(32u << 16),        // tile_dim0 = 32
                       64,                      // tile_dim1 = 64
                       ldb,                     // tensor_dim0_stride (elements)
                       0, 0};
        v4i gz = (v4i){0, 0, 0, 0};
#if defined(__clang_major__) && (__clang_major__ >= 23)
        v8i gz8 = (v8i){0, 0, 0, 0, 0, 0, 0, 0};
        __builtin_amdgcn_tensor_load_to_lds(g0, g1, gz, gz, gz8, 0);
#else
        __builtin_amdgcn_tensor_load_to_lds(g0, g1, gz, gz, 0);
#endif
        __builtin_amdgcn_s_wait_tensorcnt(0);
      }
    } else {
      // vectorized manual path (handles Nn-partial tiles with zero fill)
#pragma unroll
      for (int i = 0; i < 2; ++i) {
        int idx = tid + (i << 7);       // 0..255
        int row = idx >> 2;             // n 0..63
        int ch  = (idx & 3) << 3;       // k sub-chunk *8
        uint4 w = {0u, 0u, 0u, 0u};
        if (nBase + row < Nn)
          w = *(const uint4*)(Bt + (size_t)(nBase + row) * ldb + k0 + ch);
        *(uint4*)&sB[row][ch] = w;
      }
    }
    __syncthreads();

    // ---- fragments + WMMA --------------------------------------------------
    Frag fa[2], fb[2];
#pragma unroll
    for (int im = 0; im < 2; ++im) {
      const bf16_t* p = &sA[wm + (im << 4) + ln][0];
      fa[im].q[0] = *(const uint4*)(p + 8 * g);
      fa[im].q[1] = *(const uint4*)(p + 16 + 8 * g);
    }
#pragma unroll
    for (int in = 0; in < 2; ++in) {
      const bf16_t* p = &sB[wn + (in << 4) + ln][0];
      fb[in].q[0] = *(const uint4*)(p + 8 * g);
      fb[in].q[1] = *(const uint4*)(p + 16 + 8 * g);
    }
#pragma unroll
    for (int im = 0; im < 2; ++im)
#pragma unroll
      for (int in = 0; in < 2; ++in)
        acc[im][in] = WMMA_BF16(fa[im].v, fb[in].v, acc[im][in]);
    __syncthreads();
  }

  // ---- epilogue: element r of acc -> row = 8g+r, col = lane&15 -------------
#pragma unroll
  for (int im = 0; im < 2; ++im) {
#pragma unroll
    for (int in = 0; in < 2; ++in) {
#pragma unroll
      for (int r = 0; r < 8; ++r) {
        int row = mBase + wm + (im << 4) + 8 * g + r;
        int col = nBase + wn + (in << 4) + ln;
        if (row < M && col < Nn) {
          float vv = acc[im][in][r];
          if (bias) vv += bias[col];
          if (Cin)  vv += Cin[(size_t)row * Nn + col];
          if (postact == 1) vv = fmaxf(vv, 0.f);
          Cout[(size_t)row * Nn + col] = vv;
        }
      }
    }
  }
}

// ---------------------------------------------------------------------------
// Fused attention for one (b, h, 16-query-row tile). N=512, DH=32.
// scores = (Q K^T)/sqrt(32) + rel_bias[m]; f32 softmax in LDS; out = P V.
// One wave per block; all matmuls are WMMA bf16.
// q/k/v addressed as buf[(b*512+n)*ld + h*32 + d].
// ---------------------------------------------------------------------------
__global__ __launch_bounds__(32) void attn_kernel(
    const float* __restrict__ qb, const float* __restrict__ kb,
    const float* __restrict__ vb, int ld,
    const float* __restrict__ relp, const int* __restrict__ rel_ids,
    float* __restrict__ ob, int ldo) {
  __shared__ __align__(16) float sS[16][516];  // 33 KB

  const int lane = threadIdx.x;
  const int g = lane >> 4, ln = lane & 15;
  const int qt = blockIdx.x << 4;
  const int hh = blockIdx.y;
  const int bb = blockIdx.z;
  const int rowBase = bb << 9;  // b*512
  const int hcol = hh << 5;     // h*32

  // Q fragment (A layout, K-dim = DH = 32) straight from global f32
  Frag qa;
  {
    const float* qp = qb + (size_t)(rowBase + qt + ln) * ld + hcol;
    float4 f0 = *(const float4*)(qp + 8 * g);
    float4 f1 = *(const float4*)(qp + 8 * g + 4);
    float4 f2 = *(const float4*)(qp + 16 + 8 * g);
    float4 f3 = *(const float4*)(qp + 16 + 8 * g + 4);
    pack16(qa, f0, f1, f2, f3);
  }
  const float scale = 0.17677669529663687f;  // 1/sqrt(32)

  // scores: 32 key tiles of 16
  for (int mt = 0; mt < 32; ++mt) {
    Frag kf;  // B layout: lane col = key m, elem e -> dh = 8g+e (e<8), 16+8g+e-8 (e>=8)
    const float* kp = kb + (size_t)(rowBase + (mt << 4) + ln) * ld + hcol;
    float4 f0 = *(const float4*)(kp + 8 * g);
    float4 f1 = *(const float4*)(kp + 8 * g + 4);
    float4 f2 = *(const float4*)(kp + 16 + 8 * g);
    float4 f3 = *(const float4*)(kp + 16 + 8 * g + 4);
    pack16(kf, f0, f1, f2, f3);

    v8f s = vzero8();
    s = WMMA_BF16(qa.v, kf.v, s);

    float bl = 0.f;
    if (relp) bl = relp[rel_ids[(bb << 9) + (mt << 4) + ln] * 8 + hh];
#pragma unroll
    for (int r = 0; r < 8; ++r)
      sS[8 * g + r][(mt << 4) + ln] = s[r] * scale + bl;
  }

  // softmax: 2 lanes per row, each scans 256 via float4, combine with shfl
  {
    const int row  = lane >> 1;
    const int half = lane & 1;
    float4* s4 = (float4*)(&sS[row][0] + (half << 8));
    float mx = -3.0e38f;
    for (int j = 0; j < 64; ++j) {
      float4 f = s4[j];
      mx = fmaxf(mx, fmaxf(fmaxf(f.x, f.y), fmaxf(f.z, f.w)));
    }
    mx = fmaxf(mx, __shfl_xor(mx, 1, 32));
    float sum = 0.f;
    for (int j = 0; j < 64; ++j) {
      float4 f = s4[j];
      f.x = __expf(f.x - mx); f.y = __expf(f.y - mx);
      f.z = __expf(f.z - mx); f.w = __expf(f.w - mx);
      s4[j] = f;
      sum += (f.x + f.y) + (f.z + f.w);
    }
    sum += __shfl_xor(sum, 1, 32);
    float inv = 1.f / sum;
    for (int j = 0; j < 64; ++j) {
      float4 f = s4[j];
      f.x *= inv; f.y *= inv; f.z *= inv; f.w *= inv;
      s4[j] = f;
    }
  }

  // out = P @ V : K-dim is 512 keys, 2 column tiles of 16 (DH=32)
  v8f a0 = vzero8(), a1 = vzero8();
  for (int kk = 0; kk < 512; kk += 32) {
    Frag pa;  // A frag from LDS probabilities (f32 -> bf16)
    const float* p0 = &sS[ln][kk + 8 * g];
    float4 f0 = *(const float4*)p0;
    float4 f1 = *(const float4*)(p0 + 4);
    const float* p1 = &sS[ln][kk + 16 + 8 * g];
    float4 f2 = *(const float4*)p1;
    float4 f3 = *(const float4*)(p1 + 4);
    pack16(pa, f0, f1, f2, f3);
#pragma unroll
    for (int nb = 0; nb < 2; ++nb) {
      Frag vf;  // B frag: elem e -> key m = kk + 8g + e (+8 if e>=8), col = nb*16 + ln
#pragma unroll
      for (int e2 = 0; e2 < 16; ++e2) {
        int m = kk + (g << 3) + e2 + ((e2 >> 3) << 3);
        vf.e[e2] = (bf16_t)vb[(size_t)(rowBase + m) * ld + hcol + (nb << 4) + ln];
      }
      if (nb == 0) a0 = WMMA_BF16(pa.v, vf.v, a0);
      else         a1 = WMMA_BF16(pa.v, vf.v, a1);
    }
  }
#pragma unroll
  for (int r = 0; r < 8; ++r) {
    int orow = rowBase + qt + 8 * g + r;
    ob[(size_t)orow * ldo + hcol + ln]      = a0[r];
    ob[(size_t)orow * ldo + hcol + 16 + ln] = a1[r];
  }
}

// ---------------------------------------------------------------------------
// LayerNorm over last dim (256), optional residual. One wave per row.
// ---------------------------------------------------------------------------
__global__ __launch_bounds__(128) void ln_kernel(
    const float* __restrict__ x, const float* __restrict__ res,
    const float* __restrict__ gg, const float* __restrict__ bb,
    float* __restrict__ out) {
  const int row  = blockIdx.x * 4 + (threadIdx.x >> 5);
  const int lane = threadIdx.x & 31;
  const float* xp = x + (size_t)row * 256;
  const float* rp = res ? res + (size_t)row * 256 : nullptr;
  float v[8];
  float s = 0.f, s2 = 0.f;
#pragma unroll
  for (int i = 0; i < 8; ++i) {
    int c = lane + (i << 5);
    float t = xp[c] + (rp ? rp[c] : 0.f);
    v[i] = t; s += t; s2 += t * t;
  }
#pragma unroll
  for (int m = 1; m < 32; m <<= 1) {
    s  += __shfl_xor(s,  m, 32);
    s2 += __shfl_xor(s2, m, 32);
  }
  float mean = s * (1.f / 256.f);
  float var  = s2 * (1.f / 256.f) - mean * mean;
  float inv  = rsqrtf(var + 1e-5f);
#pragma unroll
  for (int i = 0; i < 8; ++i) {
    int c = lane + (i << 5);
    out[(size_t)row * 256 + c] = (v[i] - mean) * inv * gg[c] + bb[c];
  }
}

// out[i] = (node_types[(i/256)%512]==1 ? t1[i] : t0[i]) + (res ? res[i] : 0)
__global__ void select_kernel(const float* __restrict__ t0, const float* __restrict__ t1,
                              const int* __restrict__ nt, const float* __restrict__ res,
                              float* __restrict__ out, int total) {
  int i = blockIdx.x * blockDim.x + threadIdx.x;
  if (i < total) {
    int n = (i >> 8) & 511;
    float v = (nt[n] == 1) ? t1[i] : t0[i];
    if (res) v += res[i];
    out[i] = v;
  }
}

// transpose-convert: in[K x N] f32 -> out[N x K] bf16
__global__ void cvt_t_kernel(const float* __restrict__ in, bf16_t* __restrict__ out,
                             int K, int N) {
  int i = blockIdx.x * blockDim.x + threadIdx.x;
  if (i < K * N) {
    int k = i / N, n = i - k * N;
    out[(size_t)n * K + k] = (bf16_t)in[i];
  }
}

// ---------------------------------------------------------------------------
extern "C" void kernel_launch(void* const* d_in, const int* in_sizes, int n_in,
                              void* d_out, int out_size, void* d_ws, size_t ws_size,
                              hipStream_t stream) {
  (void)in_sizes; (void)n_in; (void)out_size; (void)ws_size;

  const float* x          = (const float*)d_in[0];
  const int*   node_types = (const int*)d_in[1];
  const int*   rel_ids    = (const int*)d_in[2];
  const float* W_in  = (const float*)d_in[3];
  const float* b_in  = (const float*)d_in[4];
  const float* Wq    = (const float*)d_in[5];
  const float* Wk    = (const float*)d_in[6];
  const float* Wv    = (const float*)d_in[7];
  const float* Wo    = (const float*)d_in[8];
  const float* bq    = (const float*)d_in[9];
  const float* bk    = (const float*)d_in[10];
  const float* bv    = (const float*)d_in[11];
  const float* bo    = (const float*)d_in[12];
  const float* relpr = (const float*)d_in[13];
  const float* Wqkv  = (const float*)d_in[14];
  const float* bqkv  = (const float*)d_in[15];
  const float* Wot   = (const float*)d_in[16];
  const float* bot   = (const float*)d_in[17];
  const float* ln1_g = (const float*)d_in[18];
  const float* ln1_b = (const float*)d_in[19];
  const float* ln2_g = (const float*)d_in[20];
  const float* ln2_b = (const float*)d_in[21];
  const float* Wf1   = (const float*)d_in[22];
  const float* bf1   = (const float*)d_in[23];
  const float* Wf2   = (const float*)d_in[24];
  const float* bf2   = (const float*)d_in[25];
  const float* Wc    = (const float*)d_in[26];
  const float* bc    = (const float*)d_in[27];

  float* out = (float*)d_out;
  float* ws  = (float*)d_ws;

  const int M = 16384;  // B*N
  const int TOT = M * 256;

  // workspace layout (float elements)
  size_t off = 0;
  float* hbuf = ws + off; off += 4194304;    // h        [16384,256]
  float* t0b  = ws + off; off += 4194304;    // temp0    [16384,256]
  float* t1b  = ws + off; off += 8388608;    // temp1 / FF chunk [16384,512]
  float* msg  = ws + off; off += 4194304;    // attention out
  float* qkv  = ws + off; off += 12582912;   // qkv [16384,768] / q|k|v for HGT
  float* h2b  = ws + off; off += 4194304;    // post-LN1 h
  bf16_t* pool = (bf16_t*)(ws + off);        // transposed bf16 weights

  bf16_t* Winp  = pool;            // [256][128]
  bf16_t* Wqp   = pool + 32768;    // 4x [256][256]
  bf16_t* Wkp   = pool + 294912;
  bf16_t* Wvp   = pool + 557056;
  bf16_t* Wop   = pool + 819200;
  bf16_t* Wqkvp = pool + 1081344;  // [768][256]
  bf16_t* Wotp  = pool + 1277952;  // [256][256]
  bf16_t* Wf1p  = pool + 1343488;  // [2048][256]
  bf16_t* Wf2p  = pool + 1867776;  // [256][2048]
  bf16_t* Wcp   = pool + 2392064;  // [10][256]

  auto cvt_t = [&](const float* src, bf16_t* dst, int K, int N) {
    int n = K * N;
    cvt_t_kernel<<<(n + 255) / 256, 256, 0, stream>>>(src, dst, K, N);
  };
  cvt_t(W_in, Winp, 128, 256);
  for (int i = 0; i < 4; ++i) {  // (l, type) pairs
    cvt_t(Wq + i * 65536, Wqp + i * 65536, 256, 256);
    cvt_t(Wk + i * 65536, Wkp + i * 65536, 256, 256);
    cvt_t(Wv + i * 65536, Wvp + i * 65536, 256, 256);
    cvt_t(Wo + i * 65536, Wop + i * 65536, 256, 256);
  }
  cvt_t(Wqkv, Wqkvp, 256, 768);
  cvt_t(Wot,  Wotp,  256, 256);
  cvt_t(Wf1,  Wf1p,  256, 2048);
  cvt_t(Wf2,  Wf2p,  2048, 256);
  cvt_t(Wc,   Wcp,   256, 10);

  auto gemm = [&](const float* A, const bf16_t* Bp, const float* bias, const float* Cin,
                  float* Cout, int Nn, int K, int ldb, int preact, int postact) {
    dim3 grid((Nn + 63) / 64, M / 64);
    gemm_kernel<<<grid, 128, 0, stream>>>(A, Bp, bias, Cin, Cout, M, Nn, K, ldb,
                                          preact, postact, /*use_tdm=*/1);
  };

  // input projection: h = x @ W_in + b_in   (K=128)
  gemm(x, Winp, b_in, nullptr, hbuf, 256, 128, 128, 0, 0);

  float* qb = qkv;
  float* kb = qkv + 4194304;
  float* vb = qkv + 8388608;

  // HGT layers
  for (int l = 0; l < 2; ++l) {
    gemm(hbuf, Wqp + (l * 2 + 0) * 65536, bq + (l * 2 + 0) * 256, nullptr, t0b, 256, 256, 256, 0, 0);
    gemm(hbuf, Wqp + (l * 2 + 1) * 65536, bq + (l * 2 + 1) * 256, nullptr, t1b, 256, 256, 256, 0, 0);
    select_kernel<<<TOT / 256, 256, 0, stream>>>(t0b, t1b, node_types, nullptr, qb, TOT);

    gemm(hbuf, Wkp + (l * 2 + 0) * 65536, bk + (l * 2 + 0) * 256, nullptr, t0b, 256, 256, 256, 0, 0);
    gemm(hbuf, Wkp + (l * 2 + 1) * 65536, bk + (l * 2 + 1) * 256, nullptr, t1b, 256, 256, 256, 0, 0);
    select_kernel<<<TOT / 256, 256, 0, stream>>>(t0b, t1b, node_types, nullptr, kb, TOT);

    gemm(hbuf, Wvp + (l * 2 + 0) * 65536, bv + (l * 2 + 0) * 256, nullptr, t0b, 256, 256, 256, 0, 0);
    gemm(hbuf, Wvp + (l * 2 + 1) * 65536, bv + (l * 2 + 1) * 256, nullptr, t1b, 256, 256, 256, 0, 0);
    select_kernel<<<TOT / 256, 256, 0, stream>>>(t0b, t1b, node_types, nullptr, vb, TOT);

    attn_kernel<<<dim3(32, 8, 32), 32, 0, stream>>>(qb, kb, vb, 256,
                                                    relpr + l * 80 /*R*H*/, rel_ids, msg, 256);

    gemm(msg, Wop + (l * 2 + 0) * 65536, bo + (l * 2 + 0) * 256, nullptr, t0b, 256, 256, 256, 2, 0);
    gemm(msg, Wop + (l * 2 + 1) * 65536, bo + (l * 2 + 1) * 256, nullptr, t1b, 256, 256, 256, 2, 0);
    select_kernel<<<TOT / 256, 256, 0, stream>>>(t0b, t1b, node_types, hbuf, hbuf, TOT);
  }

  // encoder
  gemm(hbuf, Wqkvp, bqkv, nullptr, qkv, 768, 256, 256, 0, 0);
  attn_kernel<<<dim3(32, 8, 32), 32, 0, stream>>>(qkv, qkv + 256, qkv + 512, 768,
                                                  nullptr, nullptr, msg, 256);
  gemm(msg, Wotp, bot, nullptr, t0b, 256, 256, 256, 0, 0);
  ln_kernel<<<M / 4, 128, 0, stream>>>(t0b, hbuf, ln1_g, ln1_b, h2b);

  // FFN chunked over hidden dim (4 x 512)
  for (int c = 0; c < 4; ++c) {
    gemm(h2b, Wf1p + c * 512 * 256, bf1 + c * 512, nullptr, t1b, 512, 256, 256, 0, 1);
    gemm(t1b, Wf2p + c * 512, (c == 0) ? bf2 : nullptr,
         (c == 0) ? nullptr : t0b, t0b, 256, 512, 2048, 0, 0);
  }

  // final LN -> h region of d_out; classifier -> logits region
  float* outH = out + 163840;  // B*N*V
  ln_kernel<<<M / 4, 128, 0, stream>>>(t0b, h2b, ln2_g, ln2_b, outH);
  gemm(outH, Wcp, bc, nullptr, out, 10, 256, 256, 0, 0);
}
